// LGnet_57552561766672
// MI455X (gfx1250) — matrix operand order repaired and checked by reading, not compiled, not run on hardware
//
#include <hip/hip_runtime.h>
#include <hip/hip_bf16.h>

// ---------------------------------------------------------------------------
// LGnet on gfx1250 (MI455X).  B=256, T=100, F=128, H=512, O=128, M=512.
// Recurrent over T; per-step GEMMs use v_wmma_f32_16x16x32_f16, f32 accum.
// Weights pre-converted to f16 once (L2-resident, reused 100x); the f16
// weight tile is filled with global_load_async_to_lds_b128 (ASYNCcnt path),
// activations (f32) are converted on the fly into LDS.
// Block tile 64(M) x 64(N), K chunk 64; each wave32 computes a 16x32 C tile
// = 4 WMMAs per stage (A fragment reused across two B fragments).
// ---------------------------------------------------------------------------

typedef __attribute__((ext_vector_type(16))) _Float16 v16h;
typedef __attribute__((ext_vector_type(8)))  float    v8f;

#define B_  256
#define T_  100
#define F_  128
#define H_  512
#define O_  128
#define M_  512

#define LSTRIDE 72   // padded f16 row stride (144 B) keeps all runs 16B-aligned

// ---------------------------------------------------------------------------
// Generic WMMA GEMM:  C[256 x N] = A[256 x K](lda,f32) * Bt[N x K](ldb,f16)^T
//                     (+ bias[n])  (+ old C if beta)
// Requires: N % 64 == 0, K % 64 == 0  (true for all uses here).
// ---------------------------------------------------------------------------
__global__ __launch_bounds__(256)
void gemm_wmma_kernel(const float* __restrict__ A, int lda,
                      const _Float16* __restrict__ Bt, int ldb,
                      const float* __restrict__ bias,
                      float* __restrict__ C, int ldc,
                      int K, int beta)
{
    __shared__ _Float16 As[64 * LSTRIDE];
    __shared__ _Float16 Bs[64 * LSTRIDE];

    const int tid  = threadIdx.x;
    const int lane = tid & 31;
    const int wave = tid >> 5;
    const int wm   = wave & 3;          // 4 wave tiles along M (16 rows each)
    const int wn   = wave >> 2;         // 2 wave tiles along N (32 cols each)
    const int m0   = blockIdx.y * 64;
    const int n0   = blockIdx.x * 64;

    const int subM = wm * 16;
    const int subN = wn * 32;
    const int colC = lane & 15;                 // C/D: N = lane % 16
    const int rowHalf = (lane >> 4) * 8;        // C/D: lanes 16-31 -> M += 8

    v8f acc0, acc1;
    if (beta) {
#pragma unroll
        for (int r = 0; r < 8; ++r) {
            const size_t row = (size_t)(m0 + subM + rowHalf + r) * ldc;
            acc0[r] = C[row + n0 + subN + colC];
            acc1[r] = C[row + n0 + subN + 16 + colC];
        }
    } else {
#pragma unroll
        for (int r = 0; r < 8; ++r) { acc0[r] = 0.0f; acc1[r] = 0.0f; }
    }

    // cooperative fill: 64 rows x 64 k per tile, 16 elements per thread
    const int frow = tid >> 2;          // 0..63
    const int fkb  = (tid & 3) * 16;    // 0,16,32,48

    for (int k0 = 0; k0 < K; k0 += 64) {
        // ---- B tile: f16 weights, async DMA straight into LDS (no VGPRs) ----
        {
            const _Float16* bg = Bt + (size_t)(n0 + frow) * ldb + k0 + fkb;
            const unsigned ldsoff =
                (unsigned)(uintptr_t)(const void*)&Bs[frow * LSTRIDE + fkb];
            // IOFFSET applies to both LDS and memory addresses (ISA 10.7/15.18)
            asm volatile("global_load_async_to_lds_b128 %0, %1, off"
                         :: "v"(ldsoff), "v"(bg) : "memory");
            asm volatile("global_load_async_to_lds_b128 %0, %1, off offset:16"
                         :: "v"(ldsoff), "v"(bg) : "memory");
        }
        // ---- A tile: f32 activations -> f16 in LDS ----
        {
            const float* ag = A + (size_t)(m0 + frow) * lda + k0 + fkb;
#pragma unroll
            for (int j = 0; j < 16; ++j)
                As[frow * LSTRIDE + fkb + j] = (_Float16)ag[j];
        }
        asm volatile("s_wait_asynccnt 0x0" ::: "memory");
        __syncthreads();

#pragma unroll
        for (int kk = 0; kk < 64; kk += 32) {
            // --- A fragment (16x32 f16, contiguous k order, same as B) ---
            // lanes 0-15: row=lane,    K 0-15 (of this 32-chunk)
            // lanes 16-31: row=lane-16, K 16-31
            v16h af;
            {
                const int m    = subM + (lane & 15);
                const int koff = (lane >> 4) * 16;
                const _Float16* ap = &As[m * LSTRIDE + kk];
#pragma unroll
                for (int j = 0; j < 16; ++j) {
                    af[j]     = ap[koff + j];
                }
            }
            // --- B fragments (32x16: lanes 0-15 K=0-15, lanes 16-31 K=16-31) ---
            const int kb2 = (lane >> 4) * 16;
            v16h bf0, bf1;
            {
                const _Float16* bp0 = &Bs[(subN      + (lane & 15)) * LSTRIDE + kk + kb2];
                const _Float16* bp1 = &Bs[(subN + 16 + (lane & 15)) * LSTRIDE + kk + kb2];
#pragma unroll
                for (int j = 0; j < 16; ++j) bf0[j] = bp0[j];
#pragma unroll
                for (int j = 0; j < 16; ++j) bf1[j] = bp1[j];
            }
            acc0 = __builtin_amdgcn_wmma_f32_16x16x32_f16(
                       false, af, false, bf0, (short)0, acc0, false, false);
            acc1 = __builtin_amdgcn_wmma_f32_16x16x32_f16(
                       false, af, false, bf1, (short)0, acc1, false, false);
        }
        __syncthreads();
    }

    if (bias) {
        const float bv0 = bias[n0 + subN + colC];
        const float bv1 = bias[n0 + subN + 16 + colC];
#pragma unroll
        for (int r = 0; r < 8; ++r) { acc0[r] += bv0; acc1[r] += bv1; }
    }
#pragma unroll
    for (int r = 0; r < 8; ++r) {
        const size_t row = (size_t)(m0 + subM + rowHalf + r) * ldc;
        C[row + n0 + subN + colC]      = acc0[r];
        C[row + n0 + subN + 16 + colC] = acc1[r];
    }
}

// ---------------------------------------------------------------------------
// z / zp imputation gates for one time step. idx over B*F.
// ---------------------------------------------------------------------------
__global__ __launch_bounds__(256)
void zzp_kernel(const float* __restrict__ input, const float* __restrict__ X_mean,
                const float* __restrict__ W_gz, const float* __restrict__ b_gz,
                const float* __restrict__ W_gzp, const float* __restrict__ b_gzp,
                float* __restrict__ z, float* __restrict__ zp, int t)
{
    const int idx = blockIdx.x * blockDim.x + threadIdx.x;  // B*F
    const int b = idx >> 7, f = idx & 127;
    const int TF = T_ * F_;
    const float* base = input + (size_t)b * 6 * TF + (size_t)t * F_ + f;
    const float x     = base[0 * TF];
    const float xl    = base[1 * TF];
    const float mask  = base[2 * TF];
    const float delta = base[3 * TF];
    const float xlb   = base[4 * TF];
    const float dltb  = base[5 * TF];
    const float xm    = X_mean[t * F_ + f];
    const float dz  = expf(-fmaxf(delta * W_gz [f * F_ + f] + b_gz [f], 0.0f));
    const float dzp = expf(-fmaxf(dltb  * W_gzp[f * F_ + f] + b_gzp[f], 0.0f));
    z [idx] = mask * x + (1.0f - mask) * (dz  * xl  + (1.0f - dz)  * xm);
    zp[idx] = mask * x + (1.0f - mask) * (dzp * xlb + (1.0f - dzp) * xm);
}

// ---------------------------------------------------------------------------
// Row softmax over 512 columns, in place. One block (256 thr) per row.
// ---------------------------------------------------------------------------
__global__ __launch_bounds__(256)
void softmax512_kernel(float* __restrict__ x)
{
    __shared__ float red[256];
    const int b = blockIdx.x, tid = threadIdx.x;
    float* row = x + (size_t)b * M_;
    const float v0 = row[tid], v1 = row[tid + 256];
    red[tid] = fmaxf(v0, v1);
    __syncthreads();
    for (int s = 128; s > 0; s >>= 1) {
        if (tid < s) red[tid] = fmaxf(red[tid], red[tid + s]);
        __syncthreads();
    }
    const float mx = red[0];
    __syncthreads();
    const float e0 = expf(v0 - mx), e1 = expf(v1 - mx);
    red[tid] = e0 + e1;
    __syncthreads();
    for (int s = 128; s > 0; s >>= 1) {
        if (tid < s) red[tid] += red[tid + s];
        __syncthreads();
    }
    const float inv = 1.0f / red[0];
    row[tid]       = e0 * inv;
    row[tid + 256] = e1 * inv;
}

// ---------------------------------------------------------------------------
// LSTM pointwise cell (PyTorch gate order i,f,g,o). idx over B*H.
// ---------------------------------------------------------------------------
__global__ __launch_bounds__(256)
void lstm_cell_kernel(const float* __restrict__ gates,
                      float* __restrict__ c, float* __restrict__ h)
{
    const int idx = blockIdx.x * blockDim.x + threadIdx.x;  // B*H
    const int b = idx >> 9, j = idx & 511;
    const float* g = gates + (size_t)b * 4 * H_;
    const float gi = g[j], gf = g[H_ + j], gg = g[2 * H_ + j], go = g[3 * H_ + j];
    const float si = 1.0f / (1.0f + expf(-gi));
    const float sf = 1.0f / (1.0f + expf(-gf));
    const float so = 1.0f / (1.0f + expf(-go));
    const float cn = sf * c[idx] + si * tanhf(gg);
    c[idx] = cn;
    h[idx] = so * tanhf(cn);
}

__global__ __launch_bounds__(256)
void zero_kernel(float* __restrict__ p, int n)
{
    const int idx = blockIdx.x * blockDim.x + threadIdx.x;
    if (idx < n) p[idx] = 0.0f;
}

// f32 -> f16 weight conversion (run once at start)
__global__ __launch_bounds__(256)
void cvt_f16_kernel(const float* __restrict__ in, _Float16* __restrict__ out, int n)
{
    const int idx = blockIdx.x * blockDim.x + threadIdx.x;
    if (idx < n) out[idx] = (_Float16)in[idx];
}

// memT16[f*M + m] = (f16) memory[m*F + f]
__global__ __launch_bounds__(256)
void transpose_cvt_kernel(const float* __restrict__ mem, _Float16* __restrict__ mT)
{
    const int idx = blockIdx.x * blockDim.x + threadIdx.x;  // M*F
    const int m = idx >> 7, f = idx & 127;
    mT[(size_t)f * M_ + m] = (_Float16)mem[(size_t)m * F_ + f];
}

// ---------------------------------------------------------------------------
static inline void launch_gemm(const float* A, int lda, const _Float16* Bt, int ldb,
                               const float* bias, float* C, int ldc,
                               int N, int K, int beta, hipStream_t s)
{
    dim3 grid(N / 64, B_ / 64);
    gemm_wmma_kernel<<<grid, 256, 0, s>>>(A, lda, Bt, ldb, bias, C, ldc, K, beta);
}

extern "C" void kernel_launch(void* const* d_in, const int* in_sizes, int n_in,
                              void* d_out, int out_size, void* d_ws, size_t ws_size,
                              hipStream_t stream)
{
    const float* input  = (const float*)d_in[0];
    const float* X_mean = (const float*)d_in[1];
    const float* W_gz   = (const float*)d_in[2];
    const float* b_gz   = (const float*)d_in[3];
    const float* W_gzp  = (const float*)d_in[4];
    const float* b_gzp  = (const float*)d_in[5];
    const float* W_q    = (const float*)d_in[6];   // [F, 2F+O] = [128, 384]
    const float* b_q    = (const float*)d_in[7];
    const float* memory = (const float*)d_in[8];   // [512, 128]
    const float* W_ih   = (const float*)d_in[9];   // [2048, 128]
    const float* W_hh   = (const float*)d_in[10];  // [2048, 512]
    const float* b_ih   = (const float*)d_in[11];
    const float* b_hh   = (const float*)d_in[12];
    const float* W_fc   = (const float*)d_in[13];  // [128, 512]
    const float* b_fc   = (const float*)d_in[14];

    // ---- f32 scratch ----
    float* ws = (float*)d_ws;
    float* h   = ws;                 // B*H   = 131072
    float* c   = h  + B_ * H_;       // B*H   = 131072
    float* z   = c  + B_ * H_;       // B*F   =  32768
    float* zp  = z  + B_ * F_;       // B*F
    float* xi  = zp + B_ * F_;       // B*O
    float* ls  = xi + B_ * O_;       // B*F
    float* lg  = ls + B_ * F_;       // B*M   = 131072 (logits / attn)
    float* gd  = lg + B_ * M_;       // B*F
    float* gt  = gd + B_ * F_;       // B*4H  = 524288
    // ---- f16 weight copies (all offsets multiples of 16B) ----
    _Float16* w16   = (_Float16*)(gt + B_ * 4 * H_);
    _Float16* Wfc16 = w16;                        // 128*512  =  65536
    _Float16* Wq16  = Wfc16 + O_ * H_;            // 128*384  =  49152
    _Float16* Mem16 = Wq16  + F_ * (2*F_ + O_);   // 512*128  =  65536
    _Float16* MemT16= Mem16 + M_ * F_;            // 128*512  =  65536
    _Float16* Wih16 = MemT16 + F_ * M_;           // 2048*128 = 262144
    _Float16* Whh16 = Wih16 + 4*H_ * F_;          // 2048*512 = 1048576

    // init recurrent state + one-time f16 weight conversion
    zero_kernel<<<(2 * B_ * H_) / 256, 256, 0, stream>>>(h, 2 * B_ * H_);
    cvt_f16_kernel<<<(O_*H_) / 256, 256, 0, stream>>>(W_fc, Wfc16, O_*H_);
    cvt_f16_kernel<<<(F_*(2*F_+O_)) / 256, 256, 0, stream>>>(W_q, Wq16, F_*(2*F_+O_));
    cvt_f16_kernel<<<(M_*F_) / 256, 256, 0, stream>>>(memory, Mem16, M_*F_);
    transpose_cvt_kernel<<<(M_*F_) / 256, 256, 0, stream>>>(memory, MemT16);
    cvt_f16_kernel<<<(4*H_*F_) / 256, 256, 0, stream>>>(W_ih, Wih16, 4*H_*F_);
    cvt_f16_kernel<<<(4*H_*H_) / 256, 256, 0, stream>>>(W_hh, Whh16, 4*H_*H_);

    for (int t = 0; t < T_; ++t) {
        // imputation gates
        zzp_kernel<<<(B_ * F_) / 256, 256, 0, stream>>>(
            input, X_mean, W_gz, b_gz, W_gzp, b_gzp, z, zp, t);

        // x_i = h @ W_fc^T + b_fc                       [B,128]
        launch_gemm(h, H_, Wfc16, H_, b_fc, xi, O_, O_, H_, 0, stream);

        // ls = z@Wq_z^T + zp@Wq_zp^T + x_i@Wq_xi^T + b_q [B,128]
        launch_gemm(z,  F_, Wq16,        2*F_+O_, b_q,     ls, F_, F_, F_, 0, stream);
        launch_gemm(zp, F_, Wq16 + F_,   2*F_+O_, nullptr, ls, F_, F_, F_, 1, stream);
        launch_gemm(xi, O_, Wq16 + 2*F_, 2*F_+O_, nullptr, ls, F_, F_, O_, 1, stream);

        // attention logits = ls @ memory^T              [B,512]
        launch_gemm(ls, F_, Mem16, F_, nullptr, lg, M_, M_, F_, 0, stream);
        softmax512_kernel<<<B_, 256, 0, stream>>>(lg);

        // gd = attn @ memory                            [B,128]
        launch_gemm(lg, M_, MemT16, M_, nullptr, gd, F_, F_, M_, 0, stream);

        // gates = gd@W_ih^T + b_ih + h@W_hh^T + b_hh    [B,2048]
        launch_gemm(gd, F_, Wih16, F_, b_ih, gt, 4*H_, 4*H_, F_, 0, stream);
        launch_gemm(h,  H_, Whh16, H_, b_hh, gt, 4*H_, 4*H_, H_, 1, stream);

        // pointwise LSTM cell -> new h, c
        lstm_cell_kernel<<<(B_ * H_) / 256, 256, 0, stream>>>(gt, c, h);
    }

    // final output = h @ W_fc^T + b_fc  -> d_out [B,1,O]
    launch_gemm(h, H_, Wfc16, H_, b_fc, (float*)d_out, O_, O_, H_, 0, stream);
}
